// HybridBridge_20486994002314
// MI455X (gfx1250) — matrix operand order repaired and checked by reading, not compiled, not run on hardware
//
#include <hip/hip_runtime.h>
#include <hip/hip_bf16.h>

// ---------------------------------------------------------------------------
// HybridBridge fused block for MI455X (gfx1250, wave32, WMMA bf16)
// - 32-token M-tiles: each B(weight) fragment feeds two WMMAs (halves L2
//   weight traffic vs 16-token tiles, doubles per-wave ILP).
// - Attention folded into per-batch kproj/vproj (T=2, H=2 degenerate).
// - Concat-GEMM context half folded into a per-batch bias (ctx broadcast).
// ---------------------------------------------------------------------------

typedef __bf16 bf16_t;
typedef __attribute__((ext_vector_type(16))) __bf16 v16bf;
typedef __attribute__((ext_vector_type(8)))  float  v8f;

static constexpr int Dd = 512, Cc = 256, Bb = 8, Ll = 4096;

// ---------------- workspace byte offsets ----------------
static constexpr size_t WS_GAMMAP = 0;                       // B*512 f32
static constexpr size_t WS_BETA   = WS_GAMMAP + 8*512*4;
static constexpr size_t WS_GATE   = WS_BETA   + 8*512*4;
static constexpr size_t WS_CB2    = WS_GATE   + 8*512*4;     // B*512 f32 (folded concat bias)
static constexpr size_t WS_KPROJ  = WS_CB2    + 8*512*4;     // B*2*2*512 f32
static constexpr size_t WS_VPROJ  = WS_KPROJ  + 8*4*512*4;
static constexpr size_t WS_SB     = WS_VPROJ  + 8*4*512*4;   // B*4 f32
static constexpr size_t WS_CW     = WS_SB     + 256;         // 512*512 bf16 (x-half of concat_w)
static constexpr size_t WS_W1     = WS_CW     + 512*512*2;   // 1024*2048 bf16
static constexpr size_t WS_W2     = WS_W1     + 1024*2048*2; // 512*1024 bf16

// ---------------- LDS byte offsets (dynamic, aliased by phase) -------------
static constexpr int LS_MIXIN = 0;        // bf16 [32][2048] (ph1-3; ph2 uses cols 0..511)
static constexpr int LS_XF    = 131072;   // f32  [32][512]  (x ph1; upd_relu ph2-5)
static constexpr int LS_HMIX  = 196608;   // bf16 [32][1024] (ph3-4)
static constexpr int LS_UPDA  = 0;        // f32  [32][512]  (ph4-5, over dead mixin)
static constexpr int LS_T2    = 65536;    // f32  [32][512]  (ph5, over dead mixin)
static constexpr int LS_RED   = 262144;   // f32 [512]
static constexpr int LS_SC    = 264192;   // f32 [128] scores
static constexpr int LS_AW    = 264704;   // f32 [128] attn weights
static constexpr int LS_STAT  = 265216;   // f32 [192]: mu,rs | muc,rsc | muo,rso (32 each)
static constexpr int LDS_BYTES = 265984;

// ---------------------------------------------------------------------------
__global__ __launch_bounds__(256)
void hb_cvt_bf16_kernel(const float* __restrict__ src, bf16_t* __restrict__ dst, int n) {
  int i = blockIdx.x * blockDim.x + threadIdx.x;
  int stride = gridDim.x * blockDim.x;
  for (; i < n; i += stride) dst[i] = (bf16_t)src[i];
}

// concat_w is (512 x 768); keep only the x-half (cols 0..511) as bf16 (512x512)
__global__ __launch_bounds__(256)
void hb_cvt_concat_kernel(const float* __restrict__ src, bf16_t* __restrict__ dst) {
  int i = blockIdx.x * blockDim.x + threadIdx.x;
  int stride = gridDim.x * blockDim.x;
  for (; i < 512 * 512; i += stride)
    dst[i] = (bf16_t)src[(size_t)(i >> 9) * 768 + (i & 511)];
}

// ---------------------------------------------------------------------------
// Per-batch precompute: FiLM gamma/beta, gate, folded concat bias, ctx MLP ->
// ctx_tok -> k,v -> folded kproj/vproj/sb.
// ---------------------------------------------------------------------------
__global__ __launch_bounds__(256)
void hb_batch_kernel(const float* __restrict__ context,
                     const float* __restrict__ film_w, const float* __restrict__ film_b,
                     const float* __restrict__ concat_w, const float* __restrict__ concat_b,
                     const float* __restrict__ ctx_w1, const float* __restrict__ ctx_b1,
                     const float* __restrict__ ctx_w2, const float* __restrict__ ctx_b2,
                     const float* __restrict__ in_w,   const float* __restrict__ in_b,
                     const float* __restrict__ out_w,
                     const float* __restrict__ gate_w, const float* __restrict__ gate_b,
                     float* __restrict__ gammap, float* __restrict__ betaw,
                     float* __restrict__ gatew,  float* __restrict__ cb2,
                     float* __restrict__ kproj,  float* __restrict__ vproj,
                     float* __restrict__ sbv) {
  __shared__ float sctx[256];
  __shared__ float sh[512];
  __shared__ float sctok[1024];
  __shared__ float skv[2048];          // k[2][512] | v[2][512]
  const int tid = threadIdx.x;
  const int b   = blockIdx.x;

  sctx[tid] = context[b * Cc + tid];
  __syncthreads();

  for (int o = tid; o < 1024; o += 256) {          // FiLM: gb = ctx @ film_w.T
    float s = film_b[o];
    const float* wr = film_w + (size_t)o * 256;
    for (int i = 0; i < 256; ++i) s = fmaf(sctx[i], wr[i], s);
    if (o < 512) gammap[b * 512 + o] = 1.f + s;    // film_gain = 1
    else         betaw[b * 512 + (o - 512)] = s;
  }
  for (int o = tid; o < 512; o += 256) {           // gate = sigmoid(ctx @ gate_w.T)
    float s = gate_b[o];
    const float* wr = gate_w + (size_t)o * 256;
    for (int i = 0; i < 256; ++i) s = fmaf(sctx[i], wr[i], s);
    gatew[b * 512 + o] = 1.f / (1.f + __expf(-s));
  }
  for (int o = tid; o < 512; o += 256) {           // folded concat bias:
    float s = concat_b[o];                         // concat_b + ctx @ concat_w[:,512:].T
    const float* wr = concat_w + (size_t)o * 768 + 512;
    for (int i = 0; i < 256; ++i) s = fmaf(sctx[i], wr[i], s);
    cb2[b * 512 + o] = s;
  }
  for (int o = tid; o < 512; o += 256) {           // h = gelu(ctx @ ctx_w1.T), exact
    float s = ctx_b1[o];
    const float* wr = ctx_w1 + (size_t)o * 256;
    for (int i = 0; i < 256; ++i) s = fmaf(sctx[i], wr[i], s);
    sh[o] = 0.5f * s * (1.f + erff(s * 0.70710678118654752f));
  }
  __syncthreads();
  for (int o = tid; o < 1024; o += 256) {          // ctx_tok = h @ ctx_w2.T
    float s = ctx_b2[o];
    const float* wr = ctx_w2 + (size_t)o * 512;
    for (int i = 0; i < 512; ++i) s = fmaf(sh[i], wr[i], s);
    sctok[o] = s;
  }
  __syncthreads();
  for (int o = tid; o < 2048; o += 256) {          // k = tok@wk.T, v = tok@wv.T
    const int half = o >> 10;                      // 0 = k, 1 = v
    const int rem  = o & 1023;
    const int t = rem >> 9, j = rem & 511;
    const int row = 512 + half * 512 + j;          // wk rows 512..1023, wv 1024..1535
    float s = in_b[row];
    const float* wr = in_w + (size_t)row * 512;
    const float* ct = sctok + t * 512;
    for (int i = 0; i < 512; ++i) s = fmaf(ct[i], wr[i], s);
    skv[o] = s;
  }
  __syncthreads();
  for (int o = tid; o < 4096; o += 256) {          // folded projections
    const int which = o >> 11;                     // 0 = kproj, 1 = vproj
    const int rem = o & 2047;
    const int hh = rem >> 10, t = (rem >> 9) & 1, d = rem & 511;
    float s = 0.f;
    if (which == 0) {                              // kproj[d] = sum_hd wq[h*256+hd,d]*k
      const float* kk = skv + t * 512 + hh * 256;
      for (int hd = 0; hd < 256; ++hd)
        s = fmaf(in_w[(size_t)(hh * 256 + hd) * 512 + d], kk[hd], s);
      kproj[(((size_t)b * 2 + hh) * 2 + t) * 512 + d] = s;
    } else {                                       // vproj[d] = sum_hd Wout[d,h*256+hd]*v
      const float* vv = skv + 1024 + t * 512 + hh * 256;
      const float* wr = out_w + (size_t)d * 512 + hh * 256;
      for (int hd = 0; hd < 256; ++hd) s = fmaf(wr[hd], vv[hd], s);
      vproj[(((size_t)b * 2 + hh) * 2 + t) * 512 + d] = s;
    }
  }
  if (tid < 4) {                                   // sb[h,t] = bq . k[t,h,:]
    const int hh = tid >> 1, t = tid & 1;
    float s = 0.f;
    const float* kk = skv + t * 512 + hh * 256;
    for (int hd = 0; hd < 256; ++hd) s = fmaf(in_b[hh * 256 + hd], kk[hd], s);
    sbv[b * 4 + hh * 2 + t] = s;
  }
}

// ---------------------------------------------------------------------------
// Per-wave 32xN GEMM: bf16 A(LDS, 32xK) * W(global bf16, NxK row-major).
// Two M-tiles share each B fragment (two accumulators per N-tile).
// Fragment layouts per CDNA5 ISA 7.12.2 (wave32).
// ---------------------------------------------------------------------------
template <int N, int K, int LDA, bool RELU, bool OUTBF16>
__device__ __forceinline__ void wave_gemm32(const bf16_t* __restrict__ A,
                                            const bf16_t* __restrict__ W,
                                            const float* __restrict__ bias,
                                            void* __restrict__ outp, int ldo, int tid) {
  const int wave = tid >> 5;
  const int lane = tid & 31;
  const int m    = lane & 15;            // A row (within M-tile)
  const int kh   = (lane >> 4) << 3;     // A: K half-select {0,8}
  const int nc   = lane & 15;            // B/D column
  const int kb   = (lane >> 4) << 3;     // B: K half-select {0,8}
  const int mrow = (lane >> 4) << 3;     // D: row base {0,8}
  const bf16_t* arow0 = A + (size_t)m * LDA + kh;
  const bf16_t* arow1 = A + (size_t)(m + 16) * LDA + kh;

  for (int nt = wave << 4; nt < N; nt += 8 << 4) {
    v8f acc0 = {}, acc1 = {};
    const bf16_t* wrow = W + (size_t)(nt + nc) * K + kb;
#pragma unroll 4
    for (int k0 = 0; k0 < K; k0 += 32) {
      v16bf bfv, af0, af1;
#pragma unroll
      for (int i = 0; i < 8; ++i) { bfv[i] = wrow[k0 + i]; bfv[i + 8] = wrow[k0 + 16 + i]; }
#pragma unroll
      for (int i = 0; i < 8; ++i) { af0[i] = arow0[k0 + i]; af0[i + 8] = arow0[k0 + 16 + i]; }
#pragma unroll
      for (int i = 0; i < 8; ++i) { af1[i] = arow1[k0 + i]; af1[i + 8] = arow1[k0 + 16 + i]; }
      __builtin_prefetch(wrow + k0 + 256, 0, 0);   // global_prefetch_b8 on weight stream
      acc0 = __builtin_amdgcn_wmma_f32_16x16x32_bf16(false, af0, false, bfv,
                                                     (short)0, acc0, false, false);
      acc1 = __builtin_amdgcn_wmma_f32_16x16x32_bf16(false, af1, false, bfv,
                                                     (short)0, acc1, false, false);
    }
    const float bv = bias[nt + nc];
#pragma unroll
    for (int r = 0; r < 8; ++r) {
      float v0 = acc0[r] + bv;
      float v1 = acc1[r] + bv;
      if (RELU) { v0 = fmaxf(v0, 0.f); v1 = fmaxf(v1, 0.f); }
      if (OUTBF16) {
        ((bf16_t*)outp)[(size_t)(mrow + r) * ldo + nt + nc]      = (bf16_t)v0;
        ((bf16_t*)outp)[(size_t)(mrow + r + 16) * ldo + nt + nc] = (bf16_t)v1;
      } else {
        ((float*)outp)[(size_t)(mrow + r) * ldo + nt + nc]      = v0;
        ((float*)outp)[(size_t)(mrow + r + 16) * ldo + nt + nc] = v1;
      }
    }
  }
}

// mean / rsqrt(var+eps) over 32 rows x 512 cols (all 256 threads participate)
__device__ __forceinline__ void row_stats512(const float* __restrict__ buf,
                                             float* __restrict__ red,
                                             float* __restrict__ mu,
                                             float* __restrict__ rs, int tid) {
  const int m = tid >> 3, ln = tid & 7;
  float s = 0.f, s2 = 0.f;
  const float* row = buf + m * 512;
  for (int j = ln; j < 512; j += 8) { float v = row[j]; s += v; s2 = fmaf(v, v, s2); }
  red[tid] = s; red[256 + tid] = s2;
  __syncthreads();
  if (tid < 32) {
    float ts = 0.f, tq = 0.f;
    for (int i = 0; i < 8; ++i) { ts += red[(tid << 3) + i]; tq += red[256 + (tid << 3) + i]; }
    const float mean = ts * (1.f / 512.f);
    const float var  = tq * (1.f / 512.f) - mean * mean;
    mu[tid] = mean;
    rs[tid] = rsqrtf(var + 1e-5f);
  }
  __syncthreads();
}

// ---------------------------------------------------------------------------
// Main fused kernel: one 32-token tile per block (256 threads = 8 wave32).
// ---------------------------------------------------------------------------
__global__ __launch_bounds__(256)
void hb_main_kernel(const float* __restrict__ x, const float* __restrict__ ctx,
                    const unsigned char* __restrict__ pvm,
                    const float* __restrict__ film_g, const float* __restrict__ film_bb,
                    const float* __restrict__ cln_g, const float* __restrict__ cln_b,
                    const float* __restrict__ q_g, const float* __restrict__ q_b,
                    const float* __restrict__ outp_b,
                    const float* __restrict__ mix_b1, const float* __restrict__ mix_b2,
                    const float* __restrict__ oln_g, const float* __restrict__ oln_b,
                    const float* __restrict__ gammap, const float* __restrict__ betaw,
                    const float* __restrict__ gatew,  const float* __restrict__ cb2,
                    const float* __restrict__ kproj,  const float* __restrict__ vproj,
                    const float* __restrict__ sbv,
                    const bf16_t* __restrict__ cw_bf, const bf16_t* __restrict__ w1_bf,
                    const bf16_t* __restrict__ w2_bf, float* __restrict__ out) {
  extern __shared__ char smem[];
  bf16_t* mixin = (bf16_t*)(smem + LS_MIXIN);
  float*  xf    = (float*)(smem + LS_XF);      // x in ph1; relu(concat) ph2+
  bf16_t* hmx   = (bf16_t*)(smem + LS_HMIX);
  float*  upda  = (float*)(smem + LS_UPDA);
  float*  t2b   = (float*)(smem + LS_T2);
  float*  red   = (float*)(smem + LS_RED);
  float*  scb   = (float*)(smem + LS_SC);
  float*  awb   = (float*)(smem + LS_AW);
  float*  stat  = (float*)(smem + LS_STAT);

  const int tid = threadIdx.x;
  const int b   = blockIdx.x >> 7;             // 128 tiles per batch
  const int l0  = (blockIdx.x & 127) << 5;
  const float* xg = x + ((size_t)b * Ll + l0) * Dd;

  // ---- phase 1: load x, LN stats, attention, build bf16 A-tile ----
  {
    const float4* src = (const float4*)xg;
    float4* dst = (float4*)xf;
    for (int i = tid; i < 4096; i += 256) dst[i] = src[i];
  }
  __syncthreads();
  row_stats512(xf, red, stat + 0, stat + 32, tid);

  { // scores[m,h,t] = (LNq(x) . kproj + sb)/16 ; 128 combos x 2 partials
    const int combo = tid >> 1;
    const int mm = combo >> 2, hh = (combo >> 1) & 1, tt = combo & 1;
    const int sub = tid & 1;
    const float mu = stat[mm], rs = stat[32 + mm];
    const float* kp = kproj + (((size_t)(b * 2 + hh) * 2 + tt) << 9);
    const float* xr = xf + mm * 512;
    float s = 0.f;
    const int d0 = sub << 8;
    for (int d = d0; d < d0 + 256; ++d) {
      const float lnq = fmaf((xr[d] - mu) * rs, q_g[d], q_b[d]);
      s = fmaf(lnq, kp[d], s);
    }
    red[tid] = s;
  }
  __syncthreads();
  if (tid < 128) {
    const float s = red[tid * 2] + red[tid * 2 + 1];
    const int hh = (tid >> 1) & 1, tt = tid & 1;
    scb[tid] = (s + sbv[b * 4 + hh * 2 + tt]) * (1.f / 16.f);  // /sqrt(HD=256)
  }
  __syncthreads();
  if (tid < 64) {                             // softmax over T=2 per (m,h)
    const int mm = tid >> 1, hh = tid & 1;
    const float s0 = scb[mm * 4 + hh * 2 + 0], s1 = scb[mm * 4 + hh * 2 + 1];
    const float mx = fmaxf(s0, s1);
    const float e0 = __expf(s0 - mx), e1 = __expf(s1 - mx);
    const float inv = 1.f / (e0 + e1);
    awb[mm * 4 + hh * 2 + 0] = e0 * inv;
    awb[mm * 4 + hh * 2 + 1] = e1 * inv;
  }
  __syncthreads();
  for (int idx = tid; idx < 16384; idx += 256) {  // attn_out + bf16 A-tile
    const int mm = idx >> 9, d = idx & 511;
    const float xx = xf[idx];
    float ao = outp_b[d];
    const float* vp = vproj + (((size_t)b * 4) << 9) + d;
#pragma unroll
    for (int h2 = 0; h2 < 2; ++h2)
#pragma unroll
      for (int t2 = 0; t2 < 2; ++t2)
        ao = fmaf(awb[mm * 4 + h2 * 2 + t2], vp[(size_t)(h2 * 2 + t2) << 9], ao);
    bf16_t* mrow = mixin + ((size_t)mm << 11);
    mrow[d]        = (bf16_t)xx;      // cols 0..511 double as concat-GEMM A-tile
    mrow[512 + d]  = (bf16_t)ao;
    mrow[1024 + d] = (bf16_t)(xx - ao);
    mrow[1536 + d] = (bf16_t)(xx * ao);
  }
  __syncthreads();

  // ---- phase 2: upd_relu = relu(x @ cw_x.T + cb2[b])  (WMMA bf16) ----
  // (context half of the concat GEMM is folded into per-batch bias cb2)
  wave_gemm32<512, 512, 2048, true, false>(mixin, cw_bf, cb2 + b * 512, xf, 512, tid);
  __syncthreads();
  row_stats512(xf, red, stat + 64, stat + 96, tid);   // LN stats of relu output

  // ---- phase 3: hmix = relu(mix_in @ mix_w1.T + b1)  (WMMA bf16) ----
  wave_gemm32<1024, 2048, 2048, true, true>(mixin, w1_bf, mix_b1, hmx, 1024, tid);
  __syncthreads();

  // ---- phase 4: upd_a = hmix @ mix_w2.T + b2         (WMMA bf16) ----
  wave_gemm32<512, 1024, 1024, false, false>(hmx, w2_bf, mix_b2, upda, 512, tid);
  __syncthreads();

  // ---- phase 5: out = LN(base + gate*(upd_c + upd_a)) ----
  for (int idx = tid; idx < 16384; idx += 256) {
    const int mm = idx >> 9, d = idx & 511;
    const float xx  = xg[idx];                         // re-read x (L2 hit)
    const float mu  = stat[mm],      rs  = stat[32 + mm];
    const float muc = stat[64 + mm], rsc = stat[96 + mm];
    const float lnf  = fmaf((xx - mu) * rs, film_g[d], film_bb[d]);
    const float base = fmaf(gammap[b * 512 + d], lnf, betaw[b * 512 + d]);
    const float updc = fmaf((xf[idx] - muc) * rsc, cln_g[d], cln_b[d]);
    const float g    = pvm[(size_t)b * Ll + l0 + mm] ? gatew[b * 512 + d] : 0.f;
    t2b[idx] = base + g * (updc + upda[idx]);
  }
  __syncthreads();
  row_stats512(t2b, red, stat + 128, stat + 160, tid);
  float* og = out + ((size_t)b * Ll + l0) * Dd;
  for (int idx = tid; idx < 16384; idx += 256) {
    const int mm = idx >> 9, d = idx & 511;
    og[idx] = fmaf((t2b[idx] - stat[128 + mm]) * stat[160 + mm], oln_g[d], oln_b[d]);
  }
}

// ---------------------------------------------------------------------------
extern "C" void kernel_launch(void* const* d_in, const int* in_sizes, int n_in,
                              void* d_out, int out_size, void* d_ws, size_t ws_size,
                              hipStream_t stream) {
  const float* x          = (const float*)d_in[0];
  const float* context    = (const float*)d_in[1];
  const unsigned char* pvm = (const unsigned char*)d_in[2];
  const float* film_ln_g  = (const float*)d_in[3];
  const float* film_ln_b  = (const float*)d_in[4];
  const float* film_w     = (const float*)d_in[5];
  const float* film_b     = (const float*)d_in[6];
  const float* concat_w   = (const float*)d_in[7];
  const float* concat_b   = (const float*)d_in[8];
  const float* concat_ln_g = (const float*)d_in[9];
  const float* concat_ln_b = (const float*)d_in[10];
  const float* ctx_w1     = (const float*)d_in[11];
  const float* ctx_b1     = (const float*)d_in[12];
  const float* ctx_w2     = (const float*)d_in[13];
  const float* ctx_b2     = (const float*)d_in[14];
  const float* q_ln_g     = (const float*)d_in[15];
  const float* q_ln_b     = (const float*)d_in[16];
  const float* in_proj_w  = (const float*)d_in[17];
  const float* in_proj_b  = (const float*)d_in[18];
  const float* out_proj_w = (const float*)d_in[19];
  const float* out_proj_b = (const float*)d_in[20];
  const float* mix_w1     = (const float*)d_in[21];
  const float* mix_b1     = (const float*)d_in[22];
  const float* mix_w2     = (const float*)d_in[23];
  const float* mix_b2     = (const float*)d_in[24];
  const float* out_ln_g   = (const float*)d_in[25];
  const float* out_ln_b   = (const float*)d_in[26];
  const float* gate_w     = (const float*)d_in[27];
  const float* gate_b     = (const float*)d_in[28];

  char* ws = (char*)d_ws;
  float* gammap = (float*)(ws + WS_GAMMAP);
  float* betaw  = (float*)(ws + WS_BETA);
  float* gatew  = (float*)(ws + WS_GATE);
  float* cb2    = (float*)(ws + WS_CB2);
  float* kproj  = (float*)(ws + WS_KPROJ);
  float* vproj  = (float*)(ws + WS_VPROJ);
  float* sbv    = (float*)(ws + WS_SB);
  bf16_t* cw_bf = (bf16_t*)(ws + WS_CW);
  bf16_t* w1_bf = (bf16_t*)(ws + WS_W1);
  bf16_t* w2_bf = (bf16_t*)(ws + WS_W2);

  // 1) weight conversion fp32 -> bf16 (lands in the 192MB L2, stays hot)
  hb_cvt_concat_kernel<<<512, 256, 0, stream>>>(concat_w, cw_bf);
  hb_cvt_bf16_kernel<<<2048, 256, 0, stream>>>(mix_w1, w1_bf, 1024 * 2048);
  hb_cvt_bf16_kernel<<<1024, 256, 0, stream>>>(mix_w2, w2_bf, 512 * 1024);

  // 2) per-batch precompute (folds q/out_proj GEMMs + concat ctx-half bias)
  hb_batch_kernel<<<8, 256, 0, stream>>>(context, film_w, film_b, concat_w, concat_b,
                                         ctx_w1, ctx_b1, ctx_w2, ctx_b2,
                                         in_proj_w, in_proj_b, out_proj_w,
                                         gate_w, gate_b,
                                         gammap, betaw, gatew, cb2, kproj, vproj, sbv);

  // 3) fused main kernel: 32-token tiles, 8 wave32 waves each
  hb_main_kernel<<<1024, 256, LDS_BYTES, stream>>>(
      x, context, pvm, film_ln_g, film_ln_b, concat_ln_g, concat_ln_b,
      q_ln_g, q_ln_b, out_proj_b, mix_b1, mix_b2, out_ln_g, out_ln_b,
      gammap, betaw, gatew, cb2, kproj, vproj, sbv, cw_bf, w1_bf, w2_bf,
      (float*)d_out);
}
